// residual_fit_67405216743852
// MI455X (gfx1250) — compile-verified, matrix operand order checked by reading, not run on hardware
//
#include <hip/hip_runtime.h>
#include <hip/hip_bf16.h>

// ---------------- types for WMMA ----------------
typedef __attribute__((ext_vector_type(16))) __bf16 v16bf;
typedef __attribute__((ext_vector_type(8)))  float  v8f;
typedef unsigned short u16;
typedef __attribute__((ext_vector_type(8)))  unsigned short v8us;

// problem constants
#define D_DATA 512
#define D_MODEL 1024
#define D_INNER 1024
#define D_STATE 16
#define DT_RANK 64
#define BB 2
#define TT 2032          // 2*1016
#define LZ 1016
#define ROWS (BB*TT)     // 4064
#define NXZ 2048
#define NDBC 96
#define M_RES (BB*LZ)    // 2032 (real rows of final GEMM)
#define M_RES_PAD 2048   // padded to multiple of 32

__device__ __forceinline__ u16 f2bf(float f) {
    unsigned u = __builtin_bit_cast(unsigned, f);
    unsigned r = u + 0x7FFFu + ((u >> 16) & 1u);
    return (u16)(r >> 16);
}

// Load a 16x32 bf16 WMMA A/B fragment row chunk.
// p points at row_base + k0 + 8*half (elements). Chunk2 is +16 elements.
__device__ __forceinline__ v16bf load_frag(const u16* p) {
    union U { v16bf v; v8us h[2]; } u;
    u.h[0] = *(const v8us*)(p);
    u.h[1] = *(const v8us*)(p + 16);
    return u.v;
}

// ---------------- generic WMMA GEMM: C[M,N] = A[M,K] x W[N,K]^T ----------------
// A: (>=grid-covered M) x K bf16 row-major ; Bw: N x K bf16 row-major ; C: M x N f32
// one wave per block; (16*MT) x (16*NT) tile per wave; grid = (N/(16*NT), ceil(M/(16*MT)))
// A must be allocated (and readable) for all grid-covered rows; stores guarded by gm < M.
// mode: 0 = store, 1 = softplus(v + bias[n]), 2 = v + resid[idx], 3 = -|v + bias[n]|
template<int MT, int NT>
__global__ void wmma_gemm(const u16* __restrict__ A, const u16* __restrict__ Bw,
                          float* C, int M, int N, int K,
                          const float* __restrict__ bias, const float* resid, int mode)
{
    const int lane = threadIdx.x;
    const int half = lane >> 4;
    const int mn   = lane & 15;
    const int N0 = blockIdx.x * (16 * NT);
    const int M0 = blockIdx.y * (16 * MT);

    v8f acc[MT][NT];
#pragma unroll
    for (int i = 0; i < MT; ++i)
#pragma unroll
        for (int j = 0; j < NT; ++j) acc[i][j] = (v8f){};

    const u16* arow[MT];
#pragma unroll
    for (int i = 0; i < MT; ++i)
        arow[i] = A + (size_t)(M0 + 16 * i + mn) * K + 8 * half;
    const u16* brow[NT];
#pragma unroll
    for (int j = 0; j < NT; ++j)
        brow[j] = Bw + (size_t)(N0 + 16 * j + mn) * K + 8 * half;

    for (int k0 = 0; k0 < K; k0 += 32) {
        v16bf af[MT];
#pragma unroll
        for (int i = 0; i < MT; ++i) af[i] = load_frag(arow[i] + k0);
        v16bf bf[NT];
#pragma unroll
        for (int j = 0; j < NT; ++j) bf[j] = load_frag(brow[j] + k0);
#pragma unroll
        for (int i = 0; i < MT; ++i)
#pragma unroll
            for (int j = 0; j < NT; ++j)
                acc[i][j] = __builtin_amdgcn_wmma_f32_16x16x32_bf16(
                    false, af[i], false, bf[j], (short)0, acc[i][j], false, false);
    }

    // epilogue
#pragma unroll
    for (int j = 0; j < NT; ++j) {
        const int gn = N0 + 16 * j + mn;
        float bv = 0.f;
        if (mode == 1 || mode == 3) bv = bias[gn];
#pragma unroll
        for (int i = 0; i < MT; ++i) {
#pragma unroll
            for (int r = 0; r < 8; ++r) {
                int gm = M0 + 16 * i + r + 8 * half;
                if (gm >= M) continue;
                size_t idx = (size_t)gm * N + gn;
                float v = acc[i][j][r];
                if (mode == 1) {
                    float xx = v + bv;
                    v = (xx > 20.f) ? xx : __logf(1.f + __expf(xx));
                } else if (mode == 2) {
                    v += resid[idx];
                } else if (mode == 3) {
                    v = -fabsf(v + bv);
                }
                C[idx] = v;
            }
        }
    }
}

// ---------------- f32 -> bf16 convert ----------------
__global__ void cvt_bf16(const float* __restrict__ s, u16* __restrict__ d, int n) {
    int i = blockIdx.x * blockDim.x + threadIdx.x;
    if (i < n) d[i] = f2bf(s[i]);
}

// ---------------- build h = concat(flip(z), z); z = [x[:,4:-4,:], deriv(x)] ----------------
__global__ void build_h(const float* __restrict__ x, float* __restrict__ h) {
    int idx = blockIdx.x * blockDim.x + threadIdx.x;
    if (idx >= BB * TT * D_MODEL) return;
    int d = idx & (D_MODEL - 1);
    int s = (idx >> 10) % TT;
    int b = idx / (TT * D_MODEL);
    int t = (s < LZ) ? (LZ - 1 - s) : (s - LZ);
    const float* xb = x + (size_t)b * 1024 * D_DATA;
    float v;
    if (d < D_DATA) {
        v = xb[(size_t)(t + 4) * D_DATA + d];
    } else {
        int dd = d - D_DATA;
        v = (1.f/280.f)  * xb[(size_t)(t    ) * D_DATA + dd]
          + (-4.f/105.f) * xb[(size_t)(t + 1) * D_DATA + dd]
          + (0.2f)       * xb[(size_t)(t + 2) * D_DATA + dd]
          + (-0.8f)      * xb[(size_t)(t + 3) * D_DATA + dd]
          + (0.8f)       * xb[(size_t)(t + 5) * D_DATA + dd]
          + (-0.2f)      * xb[(size_t)(t + 6) * D_DATA + dd]
          + (4.f/105.f)  * xb[(size_t)(t + 7) * D_DATA + dd]
          + (-1.f/280.f) * xb[(size_t)(t + 8) * D_DATA + dd];
    }
    h[idx] = v;
}

// ---------------- rmsnorm -> bf16 ----------------
__global__ void rmsnorm_kernel(const float* __restrict__ h, const float* __restrict__ w,
                               u16* __restrict__ u) {
    __shared__ float red[256];
    int row = blockIdx.x;
    const float* hr = h + (size_t)row * D_MODEL;
    float vals[4]; float s = 0.f;
#pragma unroll
    for (int i = 0; i < 4; ++i) { vals[i] = hr[threadIdx.x + 256 * i]; s += vals[i] * vals[i]; }
    red[threadIdx.x] = s; __syncthreads();
    for (int off = 128; off > 0; off >>= 1) {
        if ((int)threadIdx.x < off) red[threadIdx.x] += red[threadIdx.x + off];
        __syncthreads();
    }
    float inv = rsqrtf(red[0] / (float)D_MODEL + 1e-5f);
#pragma unroll
    for (int i = 0; i < 4; ++i) {
        int c = threadIdx.x + 256 * i;
        u[(size_t)row * D_MODEL + c] = f2bf(vals[i] * inv * w[c]);
    }
}

// ---------------- causal depthwise conv (K=4) + SiLU ----------------
__global__ void conv_silu(const float* __restrict__ xz, const float* __restrict__ cw,
                          const float* __restrict__ cb, float* __restrict__ xbf,
                          u16* __restrict__ xbb) {
    int idx = blockIdx.x * blockDim.x + threadIdx.x;
    if (idx >= ROWS * D_INNER) return;
    int e = idx & (D_INNER - 1);
    int row = idx >> 10;
    int t = row % TT, b = row / TT;
    const float* src = xz + (size_t)b * TT * NXZ + e;
    float acc = cb[e];
#pragma unroll
    for (int k = 0; k < 4; ++k) {
        int tt = t - 3 + k;
        if (tt >= 0) acc += cw[e * 4 + k] * src[(size_t)tt * NXZ];
    }
    float sv = acc / (1.f + __expf(-acc));
    xbf[idx] = sv;
    xbb[idx] = f2bf(sv);
}

// ---------------- slice delta part of dbc -> bf16 ----------------
__global__ void slice_delta(const float* __restrict__ dbc, u16* __restrict__ din) {
    int idx = blockIdx.x * blockDim.x + threadIdx.x;
    if (idx >= ROWS * DT_RANK) return;
    int r = idx & (DT_RANK - 1);
    int row = idx >> 6;
    din[idx] = f2bf(dbc[(size_t)row * NDBC + r]);
}

// ---------------- selective scan: lane per (b,e,n), shfl reduce over n ----------------
__global__ void scan_kernel(const float* __restrict__ delta, const float* __restrict__ dbc,
                            const float* __restrict__ xbf, const float* __restrict__ A_log,
                            float* __restrict__ y) {
    int gid = blockIdx.x * blockDim.x + threadIdx.x;   // 32768 = 2*1024*16
    int n = gid & 15;
    int e = (gid >> 4) & (D_INNER - 1);
    int b = gid >> 14;
    float Aen = -__expf(A_log[e * D_STATE + n]);
    float hst = 0.f;
    size_t rowbase = (size_t)b * TT;
    for (int t = 0; t < TT; ++t) {
        size_t row = rowbase + t;
        float dl = delta[row * D_INNER + e];
        float xv = xbf[row * D_INNER + e];
        float Bm = dbc[row * NDBC + DT_RANK + n];
        float Cm = dbc[row * NDBC + DT_RANK + D_STATE + n];
        float dA = __expf(dl * Aen);
        hst = dA * hst + dl * Bm * xv;
        float part = hst * Cm;
        part += __shfl_xor(part, 1);
        part += __shfl_xor(part, 2);
        part += __shfl_xor(part, 4);
        part += __shfl_xor(part, 8);
        if (n == 0) y[row * D_INNER + e] = part;
    }
}

// ---------------- y = (y + D*xb) * silu(z) -> bf16 ----------------
__global__ void gate_kernel(const float* __restrict__ y, const float* __restrict__ xbf,
                            const float* __restrict__ xz, const float* __restrict__ Dsk,
                            u16* __restrict__ ybf) {
    int idx = blockIdx.x * blockDim.x + threadIdx.x;
    if (idx >= ROWS * D_INNER) return;
    int e = idx & (D_INNER - 1);
    int row = idx >> 10;
    float v = y[idx] + Dsk[e] * xbf[idx];
    float z = xz[(size_t)row * NXZ + D_INNER + e];
    v *= z / (1.f + __expf(-z));
    ybf[idx] = f2bf(v);
}

// ---- extract resid_control: copy to d_out, bf16 copy (padded to 2048 rows), zero tail ----
__global__ void copy_rc(const float* __restrict__ h, float* __restrict__ out_rc,
                        u16* __restrict__ rcbf, float* __restrict__ tail) {
    int idx = blockIdx.x * blockDim.x + threadIdx.x;     // over M_RES_PAD * D_MODEL
    if (idx >= M_RES_PAD * D_MODEL) return;
    int d = idx & (D_MODEL - 1);
    int rr = idx >> 10;
    if (rr < M_RES) {
        int b = rr / LZ, tt = rr % LZ;
        float v = h[((size_t)b * TT + LZ + tt) * D_MODEL + d];
        out_rc[idx] = v;
        rcbf[idx] = f2bf(v);
    } else {
        rcbf[idx] = 0;   // zero-pad rows 2032..2047 for the padded GEMM tile
    }
    if (idx == 0) tail[0] = 0.f;
}

// =====================================================================
extern "C" void kernel_launch(void* const* d_in, const int* in_sizes, int n_in,
                              void* d_out, int out_size, void* d_ws, size_t ws_size,
                              hipStream_t stream) {
    const float* x          = (const float*)d_in[0];
    const float* norm_w     = (const float*)d_in[2];
    const float* in_proj_w  = (const float*)d_in[3];
    const float* conv_w     = (const float*)d_in[4];
    const float* conv_b     = (const float*)d_in[5];
    const float* x_proj_w   = (const float*)d_in[6];
    const float* dt_proj_w  = (const float*)d_in[7];
    const float* dt_proj_b  = (const float*)d_in[8];
    const float* A_log      = (const float*)d_in[9];
    const float* D_skip     = (const float*)d_in[10];
    const float* out_proj_w = (const float*)d_in[11];
    const float* resid_w    = (const float*)d_in[12];
    const float* resid_b    = (const float*)d_in[13];
    float* out = (float*)d_out;

    // workspace carve-out
    size_t off = 0;
    auto alloc = [&](size_t bytes) -> void* {
        void* p = (char*)d_ws + off;
        off += (bytes + 255) & ~(size_t)255;
        return p;
    };
    float* h      = (float*)alloc((size_t)ROWS * D_MODEL * 4);
    u16*   u_bf   = (u16*)  alloc((size_t)ROWS * D_MODEL * 2);
    float* xz     = (float*)alloc((size_t)ROWS * NXZ * 4);
    float* xb_f   = (float*)alloc((size_t)ROWS * D_INNER * 4);
    u16*   xb_bf  = (u16*)  alloc((size_t)ROWS * D_INNER * 2);
    float* dbc    = (float*)alloc((size_t)ROWS * NDBC * 4);
    u16*   din_bf = (u16*)  alloc((size_t)ROWS * DT_RANK * 2);
    float* deltab = (float*)alloc((size_t)ROWS * D_INNER * 4);
    float* yb     = (float*)alloc((size_t)ROWS * D_INNER * 4);
    u16*   y_bf   = (u16*)  alloc((size_t)ROWS * D_INNER * 2);
    u16*   rc_bf  = (u16*)  alloc((size_t)M_RES_PAD * D_MODEL * 2);
    u16*   w_in   = (u16*)  alloc((size_t)2 * 2048 * 1024 * 2);
    u16*   w_xp   = (u16*)  alloc((size_t)2 * NDBC * 1024 * 2);
    u16*   w_dt   = (u16*)  alloc((size_t)2 * 1024 * DT_RANK * 2);
    u16*   w_out  = (u16*)  alloc((size_t)2 * 1024 * 1024 * 2);
    u16*   w_res  = (u16*)  alloc((size_t)512 * 1024 * 2);

    auto cdiv = [](int a, int b) { return (a + b - 1) / b; };

    // weight conversions (f32 -> bf16)
    { int n = 2*2048*1024;    cvt_bf16<<<cdiv(n,256),256,0,stream>>>(in_proj_w,  w_in,  n); }
    { int n = 2*NDBC*1024;    cvt_bf16<<<cdiv(n,256),256,0,stream>>>(x_proj_w,   w_xp,  n); }
    { int n = 2*1024*DT_RANK; cvt_bf16<<<cdiv(n,256),256,0,stream>>>(dt_proj_w,  w_dt,  n); }
    { int n = 2*1024*1024;    cvt_bf16<<<cdiv(n,256),256,0,stream>>>(out_proj_w, w_out, n); }
    { int n = 512*1024;       cvt_bf16<<<cdiv(n,256),256,0,stream>>>(resid_w,    w_res, n); }

    // h = concat(flip(z), z)
    { int n = BB*TT*D_MODEL; build_h<<<cdiv(n,256),256,0,stream>>>(x, h); }

    for (int l = 0; l < 2; ++l) {
        // u = rmsnorm(h) -> bf16
        rmsnorm_kernel<<<ROWS, 256, 0, stream>>>(h, norm_w + (size_t)l * D_MODEL, u_bf);
        // xz = u @ in_proj^T  (4064 x 2048 x 1024), 32x64 tiles
        wmma_gemm<2,4><<<dim3(NXZ/64, ROWS/32), 32, 0, stream>>>(
            u_bf, w_in + (size_t)l * 2048 * 1024, xz, ROWS, NXZ, D_MODEL,
            nullptr, nullptr, 0);
        // xb = silu(dwconv(xz[:, :1024]))
        { int n = ROWS*D_INNER; conv_silu<<<cdiv(n,256),256,0,stream>>>(
              xz, conv_w + (size_t)l * D_INNER * 4, conv_b + (size_t)l * D_INNER, xb_f, xb_bf); }
        // dbc = xb @ x_proj^T  (4064 x 96 x 1024), 32x32 tiles (N=96)
        wmma_gemm<2,2><<<dim3(NDBC/32, ROWS/32), 32, 0, stream>>>(
            xb_bf, w_xp + (size_t)l * NDBC * 1024, dbc, ROWS, NDBC, D_INNER,
            nullptr, nullptr, 0);
        // delta slice -> bf16
        { int n = ROWS*DT_RANK; slice_delta<<<cdiv(n,256),256,0,stream>>>(dbc, din_bf); }
        // delta = softplus(din @ dt_proj^T + dt_b)  (4064 x 1024 x 64)
        wmma_gemm<2,4><<<dim3(D_INNER/64, ROWS/32), 32, 0, stream>>>(
            din_bf, w_dt + (size_t)l * 1024 * DT_RANK, deltab, ROWS, D_INNER, DT_RANK,
            dt_proj_b + (size_t)l * D_INNER, nullptr, 1);
        // selective scan
        scan_kernel<<<BB*D_INNER*D_STATE/256, 256, 0, stream>>>(
            deltab, dbc, xb_f, A_log + (size_t)l * D_INNER * D_STATE, yb);
        // gate: y = (y + D*xb)*silu(z) -> bf16
        { int n = ROWS*D_INNER; gate_kernel<<<cdiv(n,256),256,0,stream>>>(
              yb, xb_f, xz, D_skip + (size_t)l * D_INNER, y_bf); }
        // h += y @ out_proj^T  (4064 x 1024 x 1024, fused residual add)
        wmma_gemm<2,4><<<dim3(D_MODEL/64, ROWS/32), 32, 0, stream>>>(
            y_bf, w_out + (size_t)l * 1024 * 1024, h, ROWS, D_MODEL, D_INNER,
            nullptr, h, 2);
    }

    // resid_control -> d_out[part2] and zero-padded bf16 copy; zero int32 tail slot
    const size_t RC_OFF = (size_t)M_RES * D_DATA;            // 1,040,384
    const size_t TAIL   = RC_OFF + (size_t)M_RES * D_MODEL;  // 3,121,152
    { int n = M_RES_PAD*D_MODEL; copy_rc<<<cdiv(n,256),256,0,stream>>>(
          h, out + RC_OFF, rc_bf, out + TAIL); }

    // -|rc @ resid_w^T + resid_b| -> d_out[part1]  (2032 x 512 x 1024, M padded to 2048)
    wmma_gemm<2,4><<<dim3(D_DATA/64, M_RES_PAD/32), 32, 0, stream>>>(
        rc_bf, w_res, out, M_RES, D_DATA, D_MODEL,
        resid_b, nullptr, 3);
}
